// MultiHeadAttention_6966436954884
// MI455X (gfx1250) — matrix operand order repaired and checked by reading, not compiled, not run on hardware
//
#include <hip/hip_runtime.h>

// ---------------------------------------------------------------------------
// MI455X (gfx1250) multi-head attention forward.
// B=2, T=2048, C=1024, H=16, D=64.  Outputs: out [B,T,C] fp32  ++  attn [B,H,T,T] fp32.
// Bandwidth-bound on the 536MB attention-weight tensor -> all GEMMs in bf16 WMMA,
// fp32 only where the output dtype requires it.
// ---------------------------------------------------------------------------

typedef __attribute__((ext_vector_type(4)))  float          f32x4;
typedef __attribute__((ext_vector_type(8)))  float          v8f;
typedef __attribute__((ext_vector_type(8)))  unsigned short u16x8;
typedef __attribute__((ext_vector_type(16))) unsigned short u16x16;
typedef __attribute__((ext_vector_type(16))) __bf16         v16bf;

#define DEV __device__ __forceinline__

constexpr int kB   = 2;
constexpr int kT   = 2048;
constexpr int kC   = 1024;
constexpr int kH   = 16;
constexpr int kD   = 64;
constexpr int kM   = kB * kT;     // 4096 rows for the projection GEMMs

// float -> bf16, round-to-nearest-even (bit trick; inf-safe)
DEV unsigned short f2bf(float f) {
  unsigned u = __builtin_bit_cast(unsigned, f);
  u += 0x7FFFu + ((u >> 16) & 1u);
  return (unsigned short)(u >> 16);
}

DEV v16bf frag_from_halves(u16x8 lo, u16x8 hi) {
  u16x16 c = __builtin_shufflevector(lo, hi, 0,1,2,3,4,5,6,7,8,9,10,11,12,13,14,15);
  return __builtin_bit_cast(v16bf, c);
}
DEV v16bf frag_contig(const unsigned short* p) {           // 32B-aligned contiguous
  return __builtin_bit_cast(v16bf, *(const u16x16*)p);
}
DEV v8f wmma_bf16(v16bf a, v16bf b, v8f c) {
  return __builtin_amdgcn_wmma_f32_16x16x32_bf16(false, a, false, b, (short)0, c, false, false);
}

// ---------------------------------------------------------------------------
// Kernel 1/2/3/5: Y[M,N] = cvt(A[M,K]) @ cvt(W[K,N]) + bias.  BM=BN=128, BK=32.
// 8 waves: 2 (M) x 4 (N); each wave 64x32 = 4x2 16x16 tiles -> 8 WMMA per k-step.
// AT in {float, unsigned short(bf16)}, OT in {float, unsigned short(bf16)}.
// ---------------------------------------------------------------------------
template <typename AT, typename OT>
__global__ __launch_bounds__(256)
void gemm_bias(const AT* __restrict__ A, const float* __restrict__ W,
               const float* __restrict__ bias, OT* __restrict__ Y,
               int M, int N, int K) {
  constexpr int BM = 128, BN = 128, BK = 32;
  __shared__ alignas(64) unsigned short As[BM][BK];
  __shared__ alignas(64) unsigned short Bs[BN][BK];   // transposed: Bs[n][k]

  const int tid  = threadIdx.x;
  const int wid  = tid >> 5, lane = tid & 31;
  const int wm   = wid & 1, wn = wid >> 1;
  const int lrow = lane & 15, half = lane >> 4;
  const size_t m0 = (size_t)blockIdx.y * BM;
  const size_t n0 = (size_t)blockIdx.x * BN;

  v8f acc[4][2] = {};

  for (int k0 = 0; k0 < K; k0 += BK) {
    if (k0 + BK < K)  // warm L2/WGP$ for next A tile
      __builtin_prefetch(&A[(m0 + (tid >> 1)) * K + k0 + BK], 0, 0);

    // ---- stage A tile (fp32 converts to bf16, bf16 copies raw) ----
    if constexpr (sizeof(AT) == 4) {
      #pragma unroll
      for (int g = tid; g < BM * BK / 4; g += 256) {
        int r = g >> 3, c = (g & 7) * 4;
        f32x4 v = *(const f32x4*)&A[(m0 + r) * K + k0 + c];
        As[r][c+0] = f2bf(v[0]); As[r][c+1] = f2bf(v[1]);
        As[r][c+2] = f2bf(v[2]); As[r][c+3] = f2bf(v[3]);
      }
    } else {
      #pragma unroll
      for (int g = tid; g < BM * BK / 8; g += 256) {
        int r = g >> 2, c = (g & 3) * 8;
        *(u16x8*)&As[r][c] = *(const u16x8*)&A[(m0 + r) * K + k0 + c];
      }
    }
    // ---- stage W tile transposed into Bs[n][k] ----
    #pragma unroll
    for (int g = tid; g < BK * BN / 4; g += 256) {
      int kr = g >> 5, c = (g & 31) * 4;
      f32x4 v = *(const f32x4*)&W[(size_t)(k0 + kr) * N + n0 + c];
      Bs[c+0][kr] = f2bf(v[0]); Bs[c+1][kr] = f2bf(v[1]);
      Bs[c+2][kr] = f2bf(v[2]); Bs[c+3][kr] = f2bf(v[3]);
    }
    __syncthreads();

    const int kb = half * 8;
    v16bf af[4], bfr[2];
    #pragma unroll
    for (int mt = 0; mt < 4; ++mt) {
      int row = wm * 64 + mt * 16 + lrow;
      af[mt] = frag_from_halves(*(const u16x8*)&As[row][kb],
                                *(const u16x8*)&As[row][16 + kb]);
    }
    #pragma unroll
    for (int nt = 0; nt < 2; ++nt)
      bfr[nt] = frag_from_halves(*(const u16x8*)&Bs[wn * 32 + nt * 16 + lrow][kb],
                                 *(const u16x8*)&Bs[wn * 32 + nt * 16 + lrow][16 + kb]);
    #pragma unroll
    for (int mt = 0; mt < 4; ++mt)
      #pragma unroll
      for (int nt = 0; nt < 2; ++nt)
        acc[mt][nt] = wmma_bf16(af[mt], bfr[nt], acc[mt][nt]);
    __syncthreads();
  }

  #pragma unroll
  for (int nt = 0; nt < 2; ++nt) {
    size_t col = n0 + wn * 32 + nt * 16 + lrow;
    float bv = bias[col];
    #pragma unroll
    for (int mt = 0; mt < 4; ++mt)
      #pragma unroll
      for (int r = 0; r < 8; ++r) {
        size_t row = m0 + wm * 64 + mt * 16 + half * 8 + r;
        float v = acc[mt][nt][r] + bv;
        if constexpr (sizeof(OT) == 2) Y[row * N + col] = f2bf(v);
        else                           Y[row * N + col] = v;
      }
  }
}

// ---------------------------------------------------------------------------
// scores[b,h,q,k] = (qh . kh) / 8, causal-masked.  128x128 tile per block,
// whole K=64 staged once; fully-masked blocks write -inf without compute.
// ---------------------------------------------------------------------------
__global__ __launch_bounds__(256)
void attn_scores(const unsigned short* __restrict__ qp,
                 const unsigned short* __restrict__ kp,
                 float* __restrict__ attn) {
  const int bh = blockIdx.z, b = bh >> 4, h = bh & 15;
  const int q0 = blockIdx.y * 128, k0 = blockIdx.x * 128;
  const float NEG_INF = -__builtin_inff();

  if (k0 > q0 + 127) {                       // entirely above the diagonal
    f32x4 ninf = {NEG_INF, NEG_INF, NEG_INF, NEG_INF};
    #pragma unroll
    for (int g = threadIdx.x; g < 128 * 128 / 4; g += 256) {
      int r = g >> 5, c = (g & 31) * 4;
      *(f32x4*)&attn[((size_t)bh * kT + q0 + r) * kT + k0 + c] = ninf;
    }
    return;
  }

  __shared__ alignas(64) unsigned short As[128][64];   // Q rows
  __shared__ alignas(64) unsigned short Bs[128][64];   // K rows (== B^T cols)

  const int tid  = threadIdx.x;
  const int wid  = tid >> 5, lane = tid & 31;
  const int wm   = wid & 1, wn = wid >> 1;
  const int lrow = lane & 15, half = lane >> 4;

  const unsigned short* qb = qp + (size_t)b * kT * kC + h * kD;
  const unsigned short* kb_ = kp + (size_t)b * kT * kC + h * kD;
  #pragma unroll
  for (int g = tid; g < 128 * 64 / 8; g += 256) {
    int r = g >> 3, c = (g & 7) * 8;
    *(u16x8*)&As[r][c] = *(const u16x8*)&qb [(size_t)(q0 + r) * kC + c];
    *(u16x8*)&Bs[r][c] = *(const u16x8*)&kb_[(size_t)(k0 + r) * kC + c];
  }
  __syncthreads();

  v8f acc[4][2] = {};
  #pragma unroll
  for (int kk = 0; kk < 64; kk += 32) {
    const int kbo = half * 8;
    v16bf af[4], bfr[2];
    #pragma unroll
    for (int mt = 0; mt < 4; ++mt) {
      int row = wm * 64 + mt * 16 + lrow;
      af[mt] = frag_from_halves(*(const u16x8*)&As[row][kk + kbo],
                                *(const u16x8*)&As[row][kk + 16 + kbo]);
    }
    #pragma unroll
    for (int nt = 0; nt < 2; ++nt)
      bfr[nt] = frag_from_halves(*(const u16x8*)&Bs[wn * 32 + nt * 16 + lrow][kk + kbo],
                                 *(const u16x8*)&Bs[wn * 32 + nt * 16 + lrow][kk + 16 + kbo]);
    #pragma unroll
    for (int mt = 0; mt < 4; ++mt)
      #pragma unroll
      for (int nt = 0; nt < 2; ++nt)
        acc[mt][nt] = wmma_bf16(af[mt], bfr[nt], acc[mt][nt]);
  }

  #pragma unroll
  for (int nt = 0; nt < 2; ++nt) {
    int kg = k0 + wn * 32 + nt * 16 + lrow;
    #pragma unroll
    for (int mt = 0; mt < 4; ++mt)
      #pragma unroll
      for (int r = 0; r < 8; ++r) {
        int qg = q0 + wm * 64 + mt * 16 + half * 8 + r;
        float v = (kg <= qg) ? acc[mt][nt][r] * 0.125f : NEG_INF;
        attn[((size_t)bh * kT + qg) * kT + kg] = v;
      }
  }
}

// ---------------------------------------------------------------------------
// In-place softmax over each length-2048 row; one wave per row, 64 fp32/lane.
// ---------------------------------------------------------------------------
__global__ __launch_bounds__(256)
void softmax_rows(float* __restrict__ attn) {
  const int tid = threadIdx.x, wid = tid >> 5, lane = tid & 31;
  float* p = attn + ((size_t)blockIdx.x * 8 + wid) * kT;

  f32x4 v[16];
  float mx = -__builtin_inff();
  #pragma unroll
  for (int j = 0; j < 16; ++j) {
    v[j] = *(const f32x4*)&p[j * 128 + lane * 4];
    mx = fmaxf(mx, fmaxf(fmaxf(v[j][0], v[j][1]), fmaxf(v[j][2], v[j][3])));
  }
  #pragma unroll
  for (int off = 16; off >= 1; off >>= 1) mx = fmaxf(mx, __shfl_xor(mx, off, 32));

  float s = 0.f;
  #pragma unroll
  for (int j = 0; j < 16; ++j)
    #pragma unroll
    for (int e = 0; e < 4; ++e) { float ev = __expf(v[j][e] - mx); v[j][e] = ev; s += ev; }
  #pragma unroll
  for (int off = 16; off >= 1; off >>= 1) s += __shfl_xor(s, off, 32);

  float inv = 1.f / s;
  #pragma unroll
  for (int j = 0; j < 16; ++j) {
    v[j] *= inv;
    *(f32x4*)&p[j * 128 + lane * 4] = v[j];
  }
}

// ---------------------------------------------------------------------------
// ctx[b,:,h*64:+64] = attn(b,h) @ V(b,h).  K-loop trimmed causally.
// 8 waves, each owns a 16-row stripe x full D (4 n-tiles) -> 4 WMMA / k-step.
// ---------------------------------------------------------------------------
__global__ __launch_bounds__(256)
void attn_ctx(const float* __restrict__ attn, const unsigned short* __restrict__ vp,
              unsigned short* __restrict__ ctx) {
  const int m0 = blockIdx.x * 128;
  const int bh = blockIdx.y, b = bh >> 4, h = bh & 15;

  __shared__ alignas(64) unsigned short As[128][32];
  __shared__ alignas(64) unsigned short Bs[64][32];   // Bs[n][k]

  const int tid  = threadIdx.x;
  const int w    = tid >> 5, lane = tid & 31;
  const int lrow = lane & 15, half = lane >> 4;

  const float*           arow  = attn + ((size_t)bh * kT + m0) * kT;
  const unsigned short*  vbase = vp + (size_t)b * kT * kC + h * kD;

  v8f acc[4] = {};
  const int ksteps = (m0 / 32 + 4) < (kT / 32) ? (m0 / 32 + 4) : (kT / 32);

  for (int ks = 0; ks < ksteps; ++ks) {
    const int kk = ks * 32;
    #pragma unroll
    for (int g = tid; g < 128 * 32 / 4; g += 256) {
      int r = g >> 3, c = (g & 7) * 4;
      f32x4 x = *(const f32x4*)&arow[(size_t)r * kT + kk + c];
      As[r][c+0] = f2bf(x[0]); As[r][c+1] = f2bf(x[1]);
      As[r][c+2] = f2bf(x[2]); As[r][c+3] = f2bf(x[3]);
    }
    {
      int kr = tid >> 3, c = (tid & 7) * 8;
      u16x8 x = *(const u16x8*)&vbase[(size_t)(kk + kr) * kC + c];
      #pragma unroll
      for (int j = 0; j < 8; ++j) Bs[c + j][kr] = x[j];
    }
    __syncthreads();

    const int kbo = half * 8;
    const int row = w * 16 + lrow;
    v16bf a = frag_from_halves(*(const u16x8*)&As[row][kbo],
                               *(const u16x8*)&As[row][16 + kbo]);
    #pragma unroll
    for (int nt = 0; nt < 4; ++nt) {
      v16bf bfr = frag_from_halves(*(const u16x8*)&Bs[nt * 16 + lrow][kbo],
                                   *(const u16x8*)&Bs[nt * 16 + lrow][16 + kbo]);
      acc[nt] = wmma_bf16(a, bfr, acc[nt]);
    }
    __syncthreads();
  }

  unsigned short* cbase = ctx + (size_t)b * kT * kC + h * kD;
  #pragma unroll
  for (int nt = 0; nt < 4; ++nt)
    #pragma unroll
    for (int r = 0; r < 8; ++r) {
      int mg = m0 + w * 16 + half * 8 + r;
      cbase[(size_t)mg * kC + nt * 16 + lrow] = f2bf(acc[nt][r]);
    }
}

// ---------------------------------------------------------------------------
extern "C" void kernel_launch(void* const* d_in, const int* in_sizes, int n_in,
                              void* d_out, int out_size, void* d_ws, size_t ws_size,
                              hipStream_t stream) {
  const float* q  = (const float*)d_in[0];
  const float* k  = (const float*)d_in[1];
  const float* Wq = (const float*)d_in[2];
  const float* bq = (const float*)d_in[3];
  const float* Wk = (const float*)d_in[4];
  const float* bk = (const float*)d_in[5];
  const float* Wv = (const float*)d_in[6];
  const float* bv = (const float*)d_in[7];
  const float* Wo = (const float*)d_in[8];
  const float* bo = (const float*)d_in[9];
  (void)in_sizes; (void)n_in; (void)out_size; (void)ws_size;

  float* out   = (float*)d_out;
  float* attnW = out + (size_t)kM * kC;              // [B,H,T,T]

  unsigned short* qp  = (unsigned short*)d_ws;       // bf16 [4096,1024]
  unsigned short* kp  = qp + (size_t)kM * kC;
  unsigned short* vp  = kp + (size_t)kM * kC;
  unsigned short* ctx = qp;                          // qp dead after scores

  dim3 gGemm(kC / 128, kM / 128);                    // (8, 32)
  gemm_bias<float,          unsigned short><<<gGemm, 256, 0, stream>>>(q,   Wq, bq, qp,  kM, kC, kC);
  gemm_bias<float,          unsigned short><<<gGemm, 256, 0, stream>>>(k,   Wk, bk, kp,  kM, kC, kC);
  gemm_bias<unsigned short, unsigned short><<<gGemm, 256, 0, stream>>>(kp,  Wv, bv, vp,  kM, kC, kC);

  attn_scores<<<dim3(kT / 128, kT / 128, kB * kH), 256, 0, stream>>>(qp, kp, attnW);
  softmax_rows<<<dim3(kB * kH * kT / 8), 256, 0, stream>>>(attnW);
  attn_ctx<<<dim3(kT / 128, kB * kH), 256, 0, stream>>>(attnW, vp, ctx);

  gemm_bias<unsigned short, float><<<gGemm, 256, 0, stream>>>(ctx, Wo, bo, out, kM, kC, kC);
}
